// SparseTransformerBase_18743237280116
// MI455X (gfx1250) — matrix operand order repaired and checked by reading, not compiled, not run on hardware
//
#include <hip/hip_runtime.h>
#include <math.h>

// ---------------- problem constants (match reference) ----------------
#define CIN   32
#define CDIM  512
#define HEADS 8
#define HD    64
#define WSWIN 8
#define NWD   4
#define NWIN  64      // NWD^3
#define CAPW  512     // WSWIN^3
#define DFF   2048
#define NBLK  4

// ---------------- CDNA5 WMMA types ----------------
typedef __bf16 bf16;
typedef __attribute__((ext_vector_type(16))) __bf16 v16bf;
typedef __attribute__((ext_vector_type(8)))  __bf16 v8bf;
typedef __attribute__((ext_vector_type(8)))  float  v8f;
typedef __attribute__((ext_vector_type(4)))  int    v4i;
typedef v4i __attribute__((address_space(1))) v4i_g;   // global int4
typedef v4i __attribute__((address_space(3))) v4i_l;   // LDS int4

#if defined(__has_builtin)
#if __has_builtin(__builtin_amdgcn_global_load_async_to_lds_b128)
#define HAVE_ASYNC_LDS 1
#endif
#if __has_builtin(__builtin_amdgcn_s_wait_asynccnt)
#define HAVE_WAIT_ASYNC 1
#endif
#endif

__device__ inline v8f zero8() {
  v8f v;
  for (int i = 0; i < 8; ++i) v[i] = 0.f;
  return v;
}

// build a 16-element bf16 fragment from two 16-byte LDS chunks
__device__ inline v16bf ld16(const bf16* p0, const bf16* p1) {
  union { v16bf v; v8bf h[2]; } u;
  u.h[0] = *(const v8bf*)p0;
  u.h[1] = *(const v8bf*)p1;
  return u.v;
}

__device__ inline v8f wmma_bf16(v16bf a, v16bf b, v8f c) {
  // D = A(16x32 bf16) x B(32x16 bf16) + C(16x16 f32)
  return __builtin_amdgcn_wmma_f32_16x16x32_bf16(
      /*neg_a=*/false, a, /*neg_b=*/false, b,
      /*c_mod=*/(short)0, c, /*reuse_a=*/false, /*reuse_b=*/false);
}

// 16-byte global -> LDS copy. Async (ASYNCcnt) path on CDNA5 toolchains that
// expose the builtin; falls back to a plain b128 load + ds_store pair.
__device__ inline void copy16_g2l(const bf16* gsrc, bf16* ldst) {
#ifdef HAVE_ASYNC_LDS
  __builtin_amdgcn_global_load_async_to_lds_b128(
      (v4i_g*)(v4i*)gsrc, (v4i_l*)(v4i*)ldst, 0, 0);
#else
  *(v8bf*)ldst = *(const v8bf*)gsrc;
#endif
}

__device__ inline void wait_async_copies() {
#ifdef HAVE_ASYNC_LDS
#ifdef HAVE_WAIT_ASYNC
  __builtin_amdgcn_s_wait_asynccnt(0);
#else
  asm volatile("s_wait_asynccnt 0" ::: "memory");
#endif
#endif
}

__device__ inline float gelu_tanh(float x) {
  float x3 = x * x * x;
  return 0.5f * x * (1.f + tanhf(0.79788456080286535588f * (x + 0.044715f * x3)));
}

// ---------------- f32 -> bf16 bulk convert (n % 8 == 0) ----------------
__global__ __launch_bounds__(256)
void cvt_bf16_kernel(const float* __restrict__ x, bf16* __restrict__ y, int n) {
  int i = (blockIdx.x * blockDim.x + threadIdx.x) * 8;
  if (i >= n) return;
  const float4* p = (const float4*)(x + i);
  float4 a = p[0], b = p[1];
  bf16* d = y + i;
  d[0] = (bf16)a.x; d[1] = (bf16)a.y; d[2] = (bf16)a.z; d[3] = (bf16)a.w;
  d[4] = (bf16)b.x; d[5] = (bf16)b.y; d[6] = (bf16)b.z; d[7] = (bf16)b.w;
}

// ---------------- embed: h = feats @ in_w^T + in_b + pos_embed ----------------
__global__ __launch_bounds__(256)
void embed_kernel(const float* __restrict__ feats, const float* __restrict__ in_w,
                  const float* __restrict__ in_b, const int* __restrict__ coords,
                  float* __restrict__ h, int N) {
  int i = blockIdx.x * blockDim.x + threadIdx.x;
  if (i >= N * CDIM) return;
  int n = i / CDIM, c = i % CDIM;
  float acc = in_b[c];
  const float* f = feats + (size_t)n * CIN;
  const float* w = in_w + (size_t)c * CIN;
  for (int k = 0; k < CIN; ++k) acc += f[k] * w[k];
  if (c < 510) {                      // 3 axes x (85 sin + 85 cos)
    int axis = c / 170, r = c % 170;
    int j = r % 85;
    float freq = __powf(10000.f, -(float)j / 85.f);
    float ang = (float)coords[n * 4 + 1 + axis] * freq;
    acc += (r >= 85) ? __cosf(ang) : __sinf(ang);
  }
  h[i] = acc;
}

// ---------------- layernorm over last dim (512) ----------------
__global__ __launch_bounds__(256)
void ln_kernel(const float* __restrict__ x, float* __restrict__ y) {
  __shared__ float red[256];
  int n = blockIdx.x, t = threadIdx.x;
  const float* row = x + (size_t)n * CDIM;
  float v0 = row[t], v1 = row[t + 256];
  red[t] = v0 + v1;
  __syncthreads();
  for (int o = 128; o > 0; o >>= 1) {
    if (t < o) red[t] += red[t + o];
    __syncthreads();
  }
  float mean = red[0] * (1.f / CDIM);
  __syncthreads();
  float c0 = v0 - mean, c1 = v1 - mean;
  red[t] = c0 * c0 + c1 * c1;
  __syncthreads();
  for (int o = 128; o > 0; o >>= 1) {
    if (t < o) red[t] += red[t + o];
    __syncthreads();
  }
  float r = rsqrtf(red[0] * (1.f / CDIM) + 1e-6f);
  float* yr = y + (size_t)n * CDIM;
  yr[t] = c0 * r;
  yr[t + 256] = c1 * r;
}

// ---------------- GEMM: Y = act(Xb @ Wb^T + bias) (+ resid) ----------------
// Xb: N x K (bf16), Wb: M x K (bf16), Y: N x M (f32). K % 32 == 0, M % 256 == 0.
// Block tile 64x256, 8 waves each computing a 32x64 subtile (2x4 WMMA frags):
// 6 fragment loads feed 8 v_wmma per K-step. Tiles staged with async
// global->LDS b128 copies (ASYNCcnt) when available.
__global__ __launch_bounds__(256)
void gemm_bf16_kernel(const bf16* __restrict__ Xb, const bf16* __restrict__ Wb,
                      const float* __restrict__ bias, const float* __restrict__ resid,
                      float* __restrict__ Y, int N, int M, int K, int act) {
  __shared__ bf16 Xs[64][40];     // stride 40 -> 80B rows, 16B-aligned chunks
  __shared__ bf16 Ws[256][40];
  const int tid = threadIdx.x;
  const int lane = tid & 31, wave = tid >> 5;
  const int rb = blockIdx.x * 64;
  const int cb = blockIdx.y * 256;
  const int wr = wave >> 2, wc = wave & 3;   // 2x4 wave grid
  const int col = lane & 15, hi = lane >> 4;

  v8f acc[2][4];
#pragma unroll
  for (int i = 0; i < 2; ++i)
#pragma unroll
    for (int j = 0; j < 4; ++j) acc[i][j] = zero8();

  const int xr = tid >> 2, xk = (tid & 3) * 8;   // one 16B X chunk / thread

  for (int k0 = 0; k0 < K; k0 += 32) {
    { // stage X tile (64 x 32 bf16 = 4KB)
      int gr = rb + xr;
      if (gr < N) {
        copy16_g2l(Xb + (size_t)gr * K + k0 + xk, &Xs[xr][xk]);
      } else {
        for (int q = 0; q < 8; ++q) Xs[xr][xk + q] = (bf16)0.0f;
      }
    }
    { // stage W tile (256 x 32 bf16 = 16KB), 4 x 16B chunks / thread
      const bf16* wsrc = Wb + (size_t)(cb + tid) * K + k0;
#pragma unroll
      for (int c = 0; c < 4; ++c) copy16_g2l(wsrc + c * 8, &Ws[tid][c * 8]);
      if (k0 + 32 < K)
        __builtin_prefetch((const void*)(wsrc + 32), 0, 1);
    }
    wait_async_copies();
    __syncthreads();
#pragma unroll
    for (int i = 0; i < 2; ++i) {
      const int ar = wr * 32 + i * 16 + col;
      v16bf a = ld16(&Xs[ar][hi * 16], &Xs[ar][hi * 16 + 8]);
#pragma unroll
      for (int j = 0; j < 4; ++j) {
        const int br = wc * 64 + j * 16 + col;
        v16bf b = ld16(&Ws[br][hi * 16], &Ws[br][hi * 16 + 8]);
        acc[i][j] = wmma_bf16(a, b, acc[i][j]);
      }
    }
    __syncthreads();
  }

#pragma unroll
  for (int i = 0; i < 2; ++i)
#pragma unroll
    for (int j = 0; j < 4; ++j) {
      int gc = cb + wc * 64 + j * 16 + col;
      float bv = bias ? bias[gc] : 0.f;
#pragma unroll
      for (int e = 0; e < 8; ++e) {
        int gr = rb + wr * 32 + i * 16 + hi * 8 + e;
        if (gr < N) {
          float y = acc[i][j][e] + bv;
          if (act == 1) y = gelu_tanh(y);
          if (resid) y += resid[(size_t)gr * M + gc];
          Y[(size_t)gr * M + gc] = y;
        }
      }
    }
}

// ---------------- window bookkeeping ----------------
__global__ __launch_bounds__(256)
void reset_kernel(int* __restrict__ idx, int* __restrict__ cnt) {
  int i = blockIdx.x * blockDim.x + threadIdx.x;
  if (i < NWIN * CAPW) idx[i] = -1;
  if (i < NWIN) cnt[i] = 0;
}

__global__ __launch_bounds__(256)
void winassign_kernel(const int* __restrict__ coords, int* __restrict__ idx,
                      int* __restrict__ cnt, int N, int shift) {
  int n = blockIdx.x * blockDim.x + threadIdx.x;
  if (n >= N) return;
  int wx = (coords[n * 4 + 1] + shift) / WSWIN;
  int wy = (coords[n * 4 + 2] + shift) / WSWIN;
  int wz = (coords[n * 4 + 3] + shift) / WSWIN;
  int w = (wx * NWD + wy) * NWD + wz;
  int slot = atomicAdd(&cnt[w], 1);      // permutation within a window is benign
  idx[w * CAPW + slot] = n;
}

// ---------------- windowed flash attention, one (window, head) per block ----------------
__global__ __launch_bounds__(256)
void attn_kernel(const float* __restrict__ qkv, const int* __restrict__ idx,
                 float* __restrict__ out) {
  __shared__ bf16 Ks[CAPW][HD];        // 64 KB, key-major (B frags for Q@K^T)
  __shared__ bf16 Vt[HD][CAPW];        // 64 KB, d-major  (B frags for P@V)
  __shared__ float maskAdd[CAPW];      // 2 KB
  __shared__ bf16 Ps[8][16][32];       // 8 KB per-wave P relayout scratch
  const int w = blockIdx.x >> 3;
  const int head = blockIdx.x & 7;
  const int tid = threadIdx.x;
  const int lane = tid & 31, wave = tid >> 5;
  const int col = lane & 15, hi = lane >> 4;

  for (int r = tid; r < CAPW; r += 256) {
    int p = idx[w * CAPW + r];
    maskAdd[r] = (p >= 0) ? 0.f : -1e9f;
    if (p >= 0) {
      const float* kp = qkv + (size_t)p * (3 * CDIM) + CDIM + head * HD;
      const float* vp = qkv + (size_t)p * (3 * CDIM) + 2 * CDIM + head * HD;
      for (int d = 0; d < HD; ++d) {
        Ks[r][d] = (bf16)kp[d];
        Vt[d][r] = (bf16)vp[d];
      }
    } else {
      for (int d = 0; d < HD; ++d) { Ks[r][d] = (bf16)0.0f; Vt[d][r] = (bf16)0.0f; }
    }
  }
  __syncthreads();

  for (int t = wave; t < CAPW / 16; t += 8) {     // Q tiles per wave
    int qrow = t * 16 + col;
    int qp = idx[w * CAPW + qrow];
    v16bf qa0, qa1;                               // Q A-frags for D-steps 0..31, 32..63
    {
      const float* qptr = (qp >= 0) ? (qkv + (size_t)qp * (3 * CDIM) + head * HD) : nullptr;
#pragma unroll
      for (int j = 0; j < 16; ++j) {
        int dbase = hi * 16 + j;
        qa0[j] = (bf16)(qptr ? qptr[dbase] * 0.125f : 0.f);        // fold 1/sqrt(64)
        qa1[j] = (bf16)(qptr ? qptr[32 + dbase] * 0.125f : 0.f);
      }
    }
    v8f o0 = zero8(), o1 = zero8(), o2 = zero8(), o3 = zero8();
    float mrun[8], lrun[8];
#pragma unroll
    for (int i = 0; i < 8; ++i) { mrun[i] = -1e30f; lrun[i] = 0.f; }

    for (int kp2 = 0; kp2 < CAPW / 32; ++kp2) {
      const int kb0 = kp2 * 32, kb1 = kb0 + 16;
      v8f s0 = zero8(), s1 = zero8();
      {
        v16bf b00 = ld16(&Ks[kb0 + col][hi * 16], &Ks[kb0 + col][hi * 16 + 8]);
        v16bf b01 = ld16(&Ks[kb0 + col][32 + hi * 16], &Ks[kb0 + col][32 + hi * 16 + 8]);
        s0 = wmma_bf16(qa0, b00, s0);
        s0 = wmma_bf16(qa1, b01, s0);
        v16bf b10 = ld16(&Ks[kb1 + col][hi * 16], &Ks[kb1 + col][hi * 16 + 8]);
        v16bf b11 = ld16(&Ks[kb1 + col][32 + hi * 16], &Ks[kb1 + col][32 + hi * 16 + 8]);
        s1 = wmma_bf16(qa0, b10, s1);
        s1 = wmma_bf16(qa1, b11, s1);
      }
      float m0 = maskAdd[kb0 + col], m1 = maskAdd[kb1 + col];
      float scale[8];
#pragma unroll
      for (int i = 0; i < 8; ++i) {               // row max across 16 lanes (half-wave)
        float a = fmaxf(s0[i] + m0, s1[i] + m1);
        for (int off = 1; off < 16; off <<= 1) a = fmaxf(a, __shfl_xor(a, off, 32));
        float nm = fmaxf(mrun[i], a);
        scale[i] = __expf(mrun[i] - nm);
        mrun[i] = nm;
      }
#pragma unroll
      for (int i = 0; i < 8; ++i) {
        float p0 = __expf(s0[i] + m0 - mrun[i]);
        float p1 = __expf(s1[i] + m1 - mrun[i]);
        Ps[wave][hi * 8 + i][col] = (bf16)p0;       // relayout C->A via LDS
        Ps[wave][hi * 8 + i][col + 16] = (bf16)p1;
        float rs = p0 + p1;
        for (int off = 1; off < 16; off <<= 1) rs += __shfl_xor(rs, off, 32);
        lrun[i] = lrun[i] * scale[i] + rs;
      }
#pragma unroll
      for (int i = 0; i < 8; ++i) {
        o0[i] *= scale[i]; o1[i] *= scale[i]; o2[i] *= scale[i]; o3[i] *= scale[i];
      }
      v16bf pa = ld16(&Ps[wave][col][hi * 16], &Ps[wave][col][hi * 16 + 8]);
      v16bf vb0 = ld16(&Vt[0 + col][kb0 + hi * 16], &Vt[0 + col][kb0 + hi * 16 + 8]);
      v16bf vb1 = ld16(&Vt[16 + col][kb0 + hi * 16], &Vt[16 + col][kb0 + hi * 16 + 8]);
      v16bf vb2 = ld16(&Vt[32 + col][kb0 + hi * 16], &Vt[32 + col][kb0 + hi * 16 + 8]);
      v16bf vb3 = ld16(&Vt[48 + col][kb0 + hi * 16], &Vt[48 + col][kb0 + hi * 16 + 8]);
      o0 = wmma_bf16(pa, vb0, o0);
      o1 = wmma_bf16(pa, vb1, o1);
      o2 = wmma_bf16(pa, vb2, o2);
      o3 = wmma_bf16(pa, vb3, o3);
    }
#pragma unroll
    for (int i = 0; i < 8; ++i) {
      int m = hi * 8 + i;
      int prow = idx[w * CAPW + t * 16 + m];
      if (prow >= 0) {
        float inv = 1.f / lrun[i];
        float* op = out + (size_t)prow * CDIM + head * HD;
        op[0 + col] = o0[i] * inv;
        op[16 + col] = o1[i] * inv;
        op[32 + col] = o2[i] * inv;
        op[48 + col] = o3[i] * inv;
      }
    }
  }
}

// ---------------- host orchestration ----------------
extern "C" void kernel_launch(void* const* d_in, const int* in_sizes, int n_in,
                              void* d_out, int out_size, void* d_ws, size_t ws_size,
                              hipStream_t stream) {
  // JAX pytree flattening order (dict keys sorted):
  //   0: feats
  //   1 + b*8 + {0:b1, 1:b2, 2:out_b, 3:out_w, 4:qkv_b, 5:qkv_w, 6:w1, 7:w2}
  //   33: in_b, 34: in_w, 35: coords
  const float* feats = (const float*)d_in[0];
  const int N = in_sizes[0] / CIN;
  const float* in_b = (const float*)d_in[33];
  const float* in_w = (const float*)d_in[34];
  const int* coords = (const int*)d_in[35];

  char* ws = (char*)d_ws;
  size_t off = 0;
  auto carve = [&](size_t bytes) -> void* {
    void* p = ws + off;
    off = (off + bytes + 255) & ~(size_t)255;
    return p;
  };
  float* h    = (float*)carve((size_t)N * CDIM * 4);
  float* z    = (float*)carve((size_t)N * CDIM * 4);
  float* qkv  = (float*)carve((size_t)N * 3 * CDIM * 4);
  float* att  = (float*)carve((size_t)N * CDIM * 4);
  float* ff   = (float*)carve((size_t)N * DFF * 4);
  bf16* zb    = (bf16*)carve((size_t)N * CDIM * 2);
  bf16* attb  = (bf16*)carve((size_t)N * CDIM * 2);
  bf16* ffb   = (bf16*)carve((size_t)N * DFF * 2);
  bf16* wb    = (bf16*)carve((size_t)CDIM * DFF * 2);   // largest weight (512x2048)
  int* idx    = (int*)carve((size_t)NWIN * CAPW * 4);
  int* cnt    = (int*)carve((size_t)NWIN * 4);
  (void)ws_size; (void)n_in;

  auto cvt = [&](const float* src, bf16* dst, size_t n) {
    cvt_bf16_kernel<<<(unsigned)((n / 8 + 255) / 256), 256, 0, stream>>>(src, dst, (int)n);
  };

  embed_kernel<<<(N * CDIM + 255) / 256, 256, 0, stream>>>(feats, in_w, in_b, coords, h, N);

  for (int b = 0; b < NBLK; ++b) {
    const float* b1    = (const float*)d_in[1 + b * 8 + 0];
    const float* b2v   = (const float*)d_in[1 + b * 8 + 1];
    const float* out_b = (const float*)d_in[1 + b * 8 + 2];
    const float* out_w = (const float*)d_in[1 + b * 8 + 3];
    const float* qkv_b = (const float*)d_in[1 + b * 8 + 4];
    const float* qkv_w = (const float*)d_in[1 + b * 8 + 5];
    const float* w1    = (const float*)d_in[1 + b * 8 + 6];
    const float* w2    = (const float*)d_in[1 + b * 8 + 7];
    int shift = (b % 2) ? (WSWIN / 2) : 0;

    // attention half
    ln_kernel<<<N, 256, 0, stream>>>(h, z);
    cvt(z, zb, (size_t)N * CDIM);
    cvt(qkv_w, wb, (size_t)(3 * CDIM) * CDIM);
    gemm_bf16_kernel<<<dim3((N + 63) / 64, (3 * CDIM) / 256), 256, 0, stream>>>(
        zb, wb, qkv_b, nullptr, qkv, N, 3 * CDIM, CDIM, 0);
    reset_kernel<<<(NWIN * CAPW + 255) / 256, 256, 0, stream>>>(idx, cnt);
    winassign_kernel<<<(N + 255) / 256, 256, 0, stream>>>(coords, idx, cnt, N, shift);
    attn_kernel<<<NWIN * HEADS, 256, 0, stream>>>(qkv, idx, att);
    cvt(att, attb, (size_t)N * CDIM);
    cvt(out_w, wb, (size_t)CDIM * CDIM);
    gemm_bf16_kernel<<<dim3((N + 63) / 64, CDIM / 256), 256, 0, stream>>>(
        attb, wb, out_b, h, h, N, CDIM, CDIM, 0);

    // MLP half
    ln_kernel<<<N, 256, 0, stream>>>(h, z);
    cvt(z, zb, (size_t)N * CDIM);
    cvt(w1, wb, (size_t)DFF * CDIM);
    gemm_bf16_kernel<<<dim3((N + 63) / 64, DFF / 256), 256, 0, stream>>>(
        zb, wb, b1, nullptr, ff, N, DFF, CDIM, 1);
    cvt(ff, ffb, (size_t)N * DFF);
    cvt(w2, wb, (size_t)CDIM * DFF);
    gemm_bf16_kernel<<<dim3((N + 63) / 64, CDIM / 256), 256, 0, stream>>>(
        ffb, wb, b2v, h, h, N, CDIM, DFF, 0);
  }
  (void)hipMemcpyAsync(d_out, h, (size_t)N * CDIM * 4, hipMemcpyDeviceToDevice, stream);
}
